// nidLSTM_31920196944182
// MI455X (gfx1250) — matrix operand order repaired and checked by reading, not compiled, not run on hardware
//
#include <hip/hip_runtime.h>
#include <hip/hip_bf16.h>

typedef __attribute__((ext_vector_type(16))) _Float16 v16h;
typedef __attribute__((ext_vector_type(8)))  _Float16 v8h;
typedef __attribute__((ext_vector_type(8)))  float    v8f;

// Problem dims
#define NB   128
#define NT   256
#define ND   320
#define NH   512
#define NO   64
#define NSEG 32
#define NS   8
#define G4H  2048        // 4*NH
#define LEPS 1e-3f

__device__ __forceinline__ v16h mk_a16(v8h lo, v8h hi) {
    v16h a;
    #pragma unroll
    for (int i = 0; i < 8; ++i) { a[i] = lo[i]; a[8 + i] = hi[i]; }
    return a;
}

// ---------------- fp32 -> f16 conversion (weights and x) ----------------
__global__ void nid_cvt_f16(const float* __restrict__ src, _Float16* __restrict__ dst, int n) {
    int i = blockIdx.x * 256 + threadIdx.x;
    if (i < n) dst[i] = (_Float16)src[i];
}

// ---------------- gx = x @ Wx^T + bx for one segment ----------------
// gx layout: [m = t*NB + b][n], m in [0,4096), n in [0,2048)
// block: 256 threads = 8 waves; wave -> one N-tile, block covers 64 M rows (4 M-tiles)
// => each B fragment is reused by 4 WMMAs (4x less Wx re-streaming from L2)
__global__ __launch_bounds__(256) void nid_gx_kernel(
    const _Float16* __restrict__ x16, const _Float16* __restrict__ Wxh,
    const float* __restrict__ bx, float* __restrict__ gx, int s)
{
    const int lane = threadIdx.x & 31;
    const int wave = threadIdx.x >> 5;
    const int nrow = lane & 15;
    const int lk   = (lane >> 4) * 8;

    const int m0 = blockIdx.y * 64;                 // M block base (flattened t*NB+b)
    const int n  = (blockIdx.x * 8 + wave) * 16 + nrow;

    const _Float16* WxS = Wxh + (size_t)s * G4H * ND + (size_t)n * ND + (lane >> 4) * 16;

    // 4 A-row pointers for this lane (rows m0 + mt*16 + nrow)
    const _Float16* pa[4];
    #pragma unroll
    for (int mt = 0; mt < 4; ++mt) {
        const int m  = m0 + mt * 16 + nrow;
        const int tt = m >> 7;                      // timestep within segment
        const int bb = m & 127;                     // batch index
        pa[mt] = x16 + ((size_t)bb * NT + (size_t)s * NSEG + tt) * ND + 2 * lk;
    }

    v8f acc[4];
    #pragma unroll
    for (int mt = 0; mt < 4; ++mt)
        #pragma unroll
        for (int vg = 0; vg < 8; ++vg) acc[mt][vg] = 0.f;

    #pragma unroll 2
    for (int kk = 0; kk < 10; ++kk) {               // K = 320 = 10 * 32
        const int kb = kk * 32;
        const v16h bm = *(const v16h*)(WxS + kb);   // 32B aligned
        #pragma unroll
        for (int mt = 0; mt < 4; ++mt) {
            const v8h lo = *(const v8h*)(pa[mt] + kb);
            const v8h hi = *(const v8h*)(pa[mt] + kb + 8);
            acc[mt] = __builtin_amdgcn_wmma_f32_16x16x32_f16(false, mk_a16(lo, hi), false, bm,
                                                             (short)0, acc[mt], false, false);
        }
    }

    const float bv = bx[s * G4H + n];
    #pragma unroll
    for (int mt = 0; mt < 4; ++mt) {
        #pragma unroll
        for (int vg = 0; vg < 8; ++vg) {
            const int mrow = m0 + mt * 16 + lk + vg;
            gx[(size_t)mrow * G4H + n] = acc[mt][vg] + bv;
        }
    }
}

// ---------------- recurrent LSTM steps + fused FC head, one segment ----------------
// grid = 8 blocks (16 batch rows each), block = 1024 threads = 32 waves
__global__ __launch_bounds__(1024, 1) void nid_rec_kernel(
    const float* __restrict__ gx, const _Float16* __restrict__ Whh,
    const float* __restrict__ bh, const _Float16* __restrict__ Wfh,
    const float* __restrict__ bf, const float* __restrict__ lna_g,
    const float* __restrict__ lna_b, const float* __restrict__ ln_g,
    const float* __restrict__ ln_b, float* __restrict__ out,
    _Float16* __restrict__ h_state, float* __restrict__ c_state, int s)
{
    __shared__ _Float16 h16[16 * NH];      // 16 KB : current hidden state (f16, WMMA A feed)
    __shared__ float gatesS[16 * G4H];     // 128 KB: pre/post-LN gates
    __shared__ float red[16 * 64 * 2];     // 8 KB : reduction scratch

    const int tid  = threadIdx.x;
    const int lane = tid & 31;
    const int wave = tid >> 5;
    const int m0   = blockIdx.x * 16;      // batch base for this block

    const int nrow = lane & 15;
    const int lk   = (lane >> 4) * 8;

    // elementwise mapping: 64 threads per batch row, 8 contiguous j each
    const int em = tid >> 6;               // row 0..15
    const int eu = tid & 63;               // 0..63
    const int jb = eu * 8;

    float creg[8];
    if (s == 0) {
        #pragma unroll
        for (int k = 0; k < 8; ++k) { creg[k] = 0.f; h16[em * NH + jb + k] = (_Float16)0.f; }
    } else {
        #pragma unroll
        for (int k = 0; k < 8; ++k) {
            creg[k] = c_state[(size_t)(m0 + em) * NH + jb + k];
            h16[em * NH + jb + k] = h_state[(size_t)(m0 + em) * NH + jb + k];
        }
    }
    __syncthreads();

    const _Float16* WhS = Whh + (size_t)s * G4H * NH;
    const _Float16* WfS = Wfh + (size_t)s * NO * NH;
    const float*    bhS = bh + s * G4H;
    const float*    bfS = bf + s * NO;

    for (int t = 0; t < NSEG; ++t) {
        // ---- phase 1: gates = gx + bh + h @ Wh^T (WMMA, 4 N-tiles per wave) ----
        v8f acc[4];
        #pragma unroll
        for (int nt = 0; nt < 4; ++nt) {
            const int n = (wave * 4 + nt) * 16 + nrow;
            const float bv = bhS[n];
            const float* gp = gx + ((size_t)t * NB + m0 + lk) * (size_t)G4H + n;
            #pragma unroll
            for (int vg = 0; vg < 8; ++vg)
                acc[nt][vg] = gp[(size_t)vg * G4H] + bv;
        }
        #pragma unroll 2
        for (int kk = 0; kk < 16; ++kk) {          // K = 512 = 16 * 32
            const int kb = kk * 32;
            const v8h lo = *(const v8h*)&h16[nrow * NH + kb + 2 * lk];        // ds_load_b128
            const v8h hi = *(const v8h*)&h16[nrow * NH + kb + 2 * lk + 8];   // ds_load_b128
            const v16h a = mk_a16(lo, hi);
            #pragma unroll
            for (int nt = 0; nt < 4; ++nt) {
                const int n = (wave * 4 + nt) * 16 + nrow;
                const v16h bm = *(const v16h*)(WhS + (size_t)n * NH + kb + (lane >> 4) * 16);
                acc[nt] = __builtin_amdgcn_wmma_f32_16x16x32_f16(false, a, false, bm,
                                                                 (short)0, acc[nt], false, false);
            }
        }
        #pragma unroll
        for (int nt = 0; nt < 4; ++nt) {
            const int n = (wave * 4 + nt) * 16 + nrow;
            #pragma unroll
            for (int vg = 0; vg < 8; ++vg)
                gatesS[(lk + vg) * G4H + n] = acc[nt][vg];
        }
        __syncthreads();

        // ---- phase 2: layer_norm_all over H within each of 4 gate groups ----
        {
            const int cb = eu * 32;
            float sum = 0.f, sq = 0.f;
            #pragma unroll 8
            for (int k = 0; k < 32; ++k) {
                float v = gatesS[em * G4H + cb + k];
                sum += v; sq += v * v;
            }
            red[(em * 64 + eu) * 2 + 0] = sum;
            red[(em * 64 + eu) * 2 + 1] = sq;
            __syncthreads();
            const int g = eu >> 4;                  // gate group of this thread's columns
            float S0 = 0.f, S1 = 0.f;
            #pragma unroll
            for (int p = 0; p < 16; ++p) {
                S0 += red[(em * 64 + g * 16 + p) * 2 + 0];
                S1 += red[(em * 64 + g * 16 + p) * 2 + 1];
            }
            const float mu   = S0 * (1.f / NH);
            const float var  = S1 * (1.f / NH) - mu * mu;
            const float rinv = rsqrtf(var + LEPS);
            #pragma unroll 8
            for (int k = 0; k < 32; ++k) {
                const int col = cb + k;
                float v = (gatesS[em * G4H + col] - mu) * rinv;
                gatesS[em * G4H + col] = v * lna_g[col] + lna_b[col];
            }
        }
        __syncthreads();

        // ---- phase 3: cell update, cell layernorm, h = sig(o)*tanh(cn) ----
        {
            float oreg[8];
            float sum = 0.f, sq = 0.f;
            #pragma unroll
            for (int k = 0; k < 8; ++k) {
                const int j = jb + k;
                const float iv = gatesS[em * G4H + j];
                const float fv = gatesS[em * G4H + NH + j];
                const float gv = gatesS[em * G4H + 2 * NH + j];
                const float ov = gatesS[em * G4H + 3 * NH + j];
                const float ig = 1.f / (1.f + __expf(-iv));
                const float fg = 1.f / (1.f + __expf(-fv));
                const float cn = fg * creg[k] + ig * tanhf(gv);
                creg[k] = cn; oreg[k] = ov;
                sum += cn; sq += cn * cn;
            }
            red[(em * 64 + eu) * 2 + 0] = sum;
            red[(em * 64 + eu) * 2 + 1] = sq;
            __syncthreads();
            float S0 = 0.f, S1 = 0.f;
            #pragma unroll
            for (int p = 0; p < 64; ++p) {
                S0 += red[(em * 64 + p) * 2 + 0];
                S1 += red[(em * 64 + p) * 2 + 1];
            }
            const float mu   = S0 * (1.f / NH);
            const float var  = S1 * (1.f / NH) - mu * mu;
            const float rinv = rsqrtf(var + LEPS);
            #pragma unroll
            for (int k = 0; k < 8; ++k) {
                const int j = jb + k;
                const float cnn = ln_g[j] * (creg[k] - mu) * rinv + ln_b[j];
                const float og  = 1.f / (1.f + __expf(-oreg[k]));
                h16[em * NH + j] = (_Float16)(og * tanhf(cnn));
            }
        }
        __syncthreads();

        // ---- phase 4: fused FC head  y = relu(h @ Wf^T + bf)  (waves 0..3) ----
        if (wave < 4) {
            const int n = wave * 16 + nrow;
            v8f facc;
            const float bv = bfS[n];
            #pragma unroll
            for (int vg = 0; vg < 8; ++vg) facc[vg] = bv;
            #pragma unroll 2
            for (int kk = 0; kk < 16; ++kk) {
                const int kb = kk * 32;
                const v8h lo = *(const v8h*)&h16[nrow * NH + kb + 2 * lk];
                const v8h hi = *(const v8h*)&h16[nrow * NH + kb + 2 * lk + 8];
                const v16h bm = *(const v16h*)(WfS + (size_t)n * NH + kb + (lane >> 4) * 16);
                facc = __builtin_amdgcn_wmma_f32_16x16x32_f16(false, mk_a16(lo, hi), false, bm,
                                                              (short)0, facc, false, false);
            }
            const int tg = s * NSEG + t;
            #pragma unroll
            for (int vg = 0; vg < 8; ++vg) {
                const int bb = m0 + lk + vg;
                out[((size_t)bb * NT + tg) * NO + n] = fmaxf(facc[vg], 0.f);
            }
        }
        __syncthreads();
    }

    // ---- save carry state for next segment ----
    #pragma unroll
    for (int k = 0; k < 8; ++k) {
        c_state[(size_t)(m0 + em) * NH + jb + k] = creg[k];
        h_state[(size_t)(m0 + em) * NH + jb + k] = h16[em * NH + jb + k];
    }
}

// ---------------- workspace layout (bytes, all 256-aligned) ----------------
#define WS_WXH ((size_t)0)
#define WS_WHH (WS_WXH + (size_t)NS * G4H * ND * 2)          // 10,485,760
#define WS_WFH (WS_WHH + (size_t)NS * G4H * NH * 2)          // +16,777,216
#define WS_XH  (WS_WFH + (size_t)NS * NO * NH * 2)           // +524,288
#define WS_GX  (WS_XH  + (size_t)NB * NT * ND * 2)           // +20,971,520
#define WS_HST (WS_GX  + (size_t)NSEG * NB * G4H * 4)        // +33,554,432
#define WS_CST (WS_HST + (size_t)NB * NH * 2)                // +131,072
// total = WS_CST + NB*NH*4  ~= 82.7 MB

extern "C" void kernel_launch(void* const* d_in, const int* in_sizes, int n_in,
                              void* d_out, int out_size, void* d_ws, size_t ws_size,
                              hipStream_t stream) {
    const float* x     = (const float*)d_in[0];
    const float* Wx    = (const float*)d_in[1];
    const float* bx    = (const float*)d_in[2];
    const float* Wh    = (const float*)d_in[3];
    const float* bh    = (const float*)d_in[4];
    const float* Wf    = (const float*)d_in[5];
    const float* bf    = (const float*)d_in[6];
    const float* lna_g = (const float*)d_in[7];
    const float* lna_b = (const float*)d_in[8];
    const float* ln_g  = (const float*)d_in[9];
    const float* ln_b  = (const float*)d_in[10];
    float* out = (float*)d_out;

    char* ws = (char*)d_ws;
    _Float16* Wxh = (_Float16*)(ws + WS_WXH);
    _Float16* Whh = (_Float16*)(ws + WS_WHH);
    _Float16* Wfh = (_Float16*)(ws + WS_WFH);
    _Float16* x16 = (_Float16*)(ws + WS_XH);
    float*    gx  = (float*)   (ws + WS_GX);
    _Float16* hst = (_Float16*)(ws + WS_HST);
    float*    cst = (float*)   (ws + WS_CST);

    // one-time per call: weights + activations -> f16
    const int nWx = NS * G4H * ND;
    const int nWh = NS * G4H * NH;
    const int nWf = NS * NO * NH;
    const int nX  = NB * NT * ND;
    nid_cvt_f16<<<(nWx + 255) / 256, 256, 0, stream>>>(Wx, Wxh, nWx);
    nid_cvt_f16<<<(nWh + 255) / 256, 256, 0, stream>>>(Wh, Whh, nWh);
    nid_cvt_f16<<<(nWf + 255) / 256, 256, 0, stream>>>(Wf, Wfh, nWf);
    nid_cvt_f16<<<(nX  + 255) / 256, 256, 0, stream>>>(x,  x16, nX);

    for (int s = 0; s < NS; ++s) {
        // gx for this segment: M = 4096 rows (64 blocks of 4 M-tiles), N = 2048 (16 groups)
        dim3 gGrid(16, 64);
        nid_gx_kernel<<<gGrid, 256, 0, stream>>>(x16, Wxh, bx, gx, s);
        // recurrence + fused FC head: 8 blocks of 16 batch rows, 32 waves each
        nid_rec_kernel<<<8, 1024, 0, stream>>>(gx, Whh, bh, Wfh, bf,
                                               lna_g, lna_b, ln_g, ln_b,
                                               out, hst, cst, s);
    }
}
